// LightningIndexer_86406152061641
// MI455X (gfx1250) — matrix operand order repaired and checked by reading, not compiled, not run on hardware
//
#include <hip/hip_runtime.h>
#include <hip/hip_bf16.h>
#include <cstdint>
#include <cstddef>

// Problem constants (match reference)
#define B_    2
#define S_    4096
#define D_    2048
#define R_    128
#define TOPK_ 64

typedef __attribute__((ext_vector_type(16))) __bf16         v16bf;
typedef __attribute__((ext_vector_type(4)))  __bf16         v4bf;
typedef __attribute__((ext_vector_type(8)))  float          v8f;
typedef __attribute__((ext_vector_type(8)))  unsigned int   u32x8;

// LDS row stride (floats): 2048 keys + 2 pad so 8*stride mod 64 banks == 16,
// staggering the two lane-half write groups onto disjoint banks.
#define ROWLEN 2050
#define LDSB   (16 * ROWLEN * 4)

// Native f32 -> bf16 (should lower to v_cvt_pk_bf16_f32-class ops on gfx1250)
__device__ __forceinline__ v16bf cvt16(float4 f0, float4 f1, float4 f2, float4 f3) {
  v16bf t;
  t[0]  = (__bf16)f0.x; t[1]  = (__bf16)f0.y; t[2]  = (__bf16)f0.z; t[3]  = (__bf16)f0.w;
  t[4]  = (__bf16)f1.x; t[5]  = (__bf16)f1.y; t[6]  = (__bf16)f1.z; t[7]  = (__bf16)f1.w;
  t[8]  = (__bf16)f2.x; t[9]  = (__bf16)f2.y; t[10] = (__bf16)f2.z; t[11] = (__bf16)f2.w;
  t[12] = (__bf16)f3.x; t[13] = (__bf16)f3.y; t[14] = (__bf16)f3.z; t[15] = (__bf16)f3.w;
  return t;
}

// Load 16 contiguous-ish bf16 (two 16B segments) straight into a fragment.
__device__ __forceinline__ v16bf load16bf(const __bf16* p0, const __bf16* p1) {
  uint4 a = *(const uint4*)p0;   // 8 bf16
  uint4 b = *(const uint4*)p1;   // 8 bf16
  u32x8 t;
  t[0]=a.x; t[1]=a.y; t[2]=a.z; t[3]=a.w;
  t[4]=b.x; t[5]=b.y; t[6]=b.z; t[7]=b.w;
  return __builtin_bit_cast(v16bf, t);
}

// ---------------------------------------------------------------------------
// Kernel 0: one-shot Wq/Wk f32 -> bf16 conversion (kills per-wave redundant
// weight conversion in the projection kernel). 2 x 128x2048 elements.
// ---------------------------------------------------------------------------
__global__ __launch_bounds__(256)
void li_cvt_w_kernel(const float* __restrict__ Wq, const float* __restrict__ Wk,
                     __bf16* __restrict__ Wqb, __bf16* __restrict__ Wkb) {
  const int i = (blockIdx.x * 256 + threadIdx.x) * 4;    // 256 blocks cover R_*D_
  if (i < R_ * D_) {
    float4 a = *(const float4*)(Wq + i);
    float4 b = *(const float4*)(Wk + i);
    v4bf ta, tb;
    ta[0]=(__bf16)a.x; ta[1]=(__bf16)a.y; ta[2]=(__bf16)a.z; ta[3]=(__bf16)a.w;
    tb[0]=(__bf16)b.x; tb[1]=(__bf16)b.y; tb[2]=(__bf16)b.z; tb[3]=(__bf16)b.w;
    *(v4bf*)(Wqb + i) = ta;
    *(v4bf*)(Wkb + i) = tb;
  }
}

// ---------------------------------------------------------------------------
// Kernel 1: low-rank projections Qa = Q*Wq^T, Ka = K*Wk^T  (f32 A, bf16 W)
// One wave per 16-row seq tile; computes all R=128 ranks (8 accum tiles),
// looping D in chunks of 32 with v_wmma_f32_16x16x32_bf16.
// ---------------------------------------------------------------------------
__global__ __launch_bounds__(128)
void li_proj_kernel(const float* __restrict__ Q, const float* __restrict__ K,
                    const __bf16* __restrict__ Wqb, const __bf16* __restrict__ Wkb,
                    __bf16* __restrict__ Qa, __bf16* __restrict__ Ka) {
  const int wave = threadIdx.x >> 5;
  const int lane = threadIdx.x & 31;
  const int tid  = blockIdx.x * 4 + wave;          // 0..1023
  const int mat  = tid >> 9;                       // 0 -> Q/Wq, 1 -> K/Wk
  const int tile = tid & 511;                      // 0..511
  const int b    = tile >> 8;                      // S_/16 = 256 tiles per batch
  const int s0   = (tile & 255) << 4;

  const float*  __restrict__ X = mat ? K   : Q;
  const __bf16* __restrict__ W = mat ? Wkb : Wqb;
  __bf16* __restrict__ out     = mat ? Ka  : Qa;

  const int nrow = lane & 15;       // A row / B col / D col
  const int kh   = lane >> 4;       // lane-half

  const float* xrow = X + ((size_t)(b * S_ + s0 + nrow)) * D_;

  v8f acc[8];
#pragma unroll
  for (int rt = 0; rt < 8; ++rt)
#pragma unroll
    for (int j = 0; j < 8; ++j) acc[rt][j] = 0.0f;

  for (int d0 = 0; d0 < D_; d0 += 32) {
    // A fragment: 16x32 of X rows; K(e) = (e/8)*16 + kh*8 + e%8
    const float* p0 = xrow + d0 + kh * 8;
    const float* p1 = p0 + 16;
    v16bf Afrag = cvt16(*(const float4*)p0, *(const float4*)(p0 + 4),
                        *(const float4*)p1, *(const float4*)(p1 + 4));
#pragma unroll
    for (int rt = 0; rt < 8; ++rt) {
      // B fragment: 32x16 of W^T; B[k][n] = W[n][k], K(e) = (e/8)*16 + kh*8 + e%8
      const __bf16* wrow = W + ((size_t)(rt * 16 + nrow)) * D_ + d0 + kh * 8;
      v16bf Bfrag = load16bf(wrow, wrow + 16);
      acc[rt] = __builtin_amdgcn_wmma_f32_16x16x32_bf16(
          false, Afrag, false, Bfrag, (short)0, acc[rt], false, false);
    }
  }

  // Store bf16: D element (VGPR v, lane) = (M = v + 8*kh, N = nrow)
#pragma unroll
  for (int rt = 0; rt < 8; ++rt)
#pragma unroll
    for (int v = 0; v < 8; ++v) {
      const int m = v + 8 * kh;
      out[((size_t)(b * S_ + s0 + m)) * R_ + rt * 16 + nrow] = (__bf16)acc[rt][v];
    }
}

// ---------------------------------------------------------------------------
// Kernel 2: fused scores + bias + top-64.
// 512 threads (16 waves) per (batch, 16-query tile). Two halves of 2048 keys:
// waves WMMA a 16x2048 score slab into LDS, lanes stash 64 values each into
// registers, then each wave runs 64 argmax extractions for its query row.
// ---------------------------------------------------------------------------
__global__ __launch_bounds__(512)
void li_score_topk_kernel(const __bf16* __restrict__ Qa,
                          const __bf16* __restrict__ Ka,
                          int* __restrict__ outIdx, float* __restrict__ outVal) {
  extern __shared__ float lds[];                    // [16][ROWLEN]

  const int w    = threadIdx.x >> 5;                // wave 0..15
  const int lane = threadIdx.x & 31;
  const int blk  = blockIdx.x;                      // 0..511
  const int b    = blk >> 8;
  const int q0   = (blk & 255) << 4;

  const int nrow = lane & 15;
  const int kh   = lane >> 4;

  const float NEG = -__builtin_inff();
  const float scale = 0.08838834764831845f;         // 1/sqrt(128)

  // A fragments: Qa 16x128 bf16, 4 chunks of K=32 (reused for all key tiles)
  v16bf Afrag[4];
  {
    const __bf16* qrow = Qa + ((size_t)(b * S_ + q0 + nrow)) * R_;
#pragma unroll
    for (int rk = 0; rk < 4; ++rk) {
      const __bf16* p0 = qrow + rk * 32 + kh * 8;
      Afrag[rk] = load16bf(p0, p0 + 16);
    }
  }

  float val[128];

#pragma unroll
  for (int half = 0; half < 2; ++half) {
    const int kbase = half * 2048 + w * 128;        // this wave's 128-key slice
#pragma unroll
    for (int t = 0; t < 8; ++t) {
      const int n0 = kbase + t * 16;
      const __bf16* krow = Ka + ((size_t)(b * S_ + n0 + nrow)) * R_;
      v8f acc;
#pragma unroll
      for (int j = 0; j < 8; ++j) acc[j] = 0.0f;
#pragma unroll
      for (int rk = 0; rk < 4; ++rk) {
        const __bf16* p = krow + rk * 32 + kh * 8;    // two 8-bf16 runs, 16 apart
        v16bf Bfrag = load16bf(p, p + 16);
        acc = __builtin_amdgcn_wmma_f32_16x16x32_bf16(
            false, Afrag[rk], false, Bfrag, (short)0, acc, false, false);
      }
      // bias + scale, stash into LDS: element (query M = v+8*kh, key = n0+nrow)
      const int key  = n0 + nrow;
      const int lcol = (n0 - half * 2048) + nrow;
#pragma unroll
      for (int v = 0; v < 8; ++v) {
        const int mq = q0 + v + 8 * kh;
        const float dist = fabsf((float)(mq - key));
        const float sc = acc[v] * scale + 0.1f * __expf(dist * (-1.0f / 256.0f));
        lds[(v + 8 * kh) * ROWLEN + lcol] = sc;
      }
    }
    __syncthreads();
    // wave w grabs its query row: val[half*64 + i] <- key (half*2048 + i*32 + lane)
#pragma unroll
    for (int i = 0; i < 64; ++i)
      val[half * 64 + i] = lds[w * ROWLEN + i * 32 + lane];
    __syncthreads();
  }

  // Iterative top-64 extraction, register-resident.
  int* oI   = outIdx + ((size_t)(b * S_ + q0 + w)) * TOPK_;
  float* oV = outVal + ((size_t)(b * S_ + q0 + w)) * TOPK_;

  int wprev = 0x7FFFFFFF;                            // no key matches initially
  for (int it = 0; it < TOPK_; ++it) {
    const int tb = wprev - lane;                     // invalidate prev winner
    float m = NEG;
    int   mk = -1;
#pragma unroll
    for (int i = 0; i < 128; ++i) {
      const int base = (i < 64) ? (i * 32) : (2048 + (i - 64) * 32);
      if (tb == base) val[i] = NEG;
      const bool g = val[i] > m;
      mk = g ? (base + lane) : mk;
      m  = g ? val[i] : m;
    }
    // wave argmax reduction (tie-break: lower index, matching lax.top_k)
#pragma unroll
    for (int off = 16; off > 0; off >>= 1) {
      const float om  = __shfl_xor(m, off, 32);
      const int   omk = __shfl_xor(mk, off, 32);
      const bool take = (om > m) || (om == m && omk < mk);
      m  = take ? om  : m;
      mk = take ? omk : mk;
    }
    wprev = mk;
    if (lane == 0) { oI[it] = mk; oV[it] = m; }
  }
}

// ---------------------------------------------------------------------------
extern "C" void kernel_launch(void* const* d_in, const int* in_sizes, int n_in,
                              void* d_out, int out_size, void* d_ws, size_t ws_size,
                              hipStream_t stream) {
  (void)in_sizes; (void)n_in; (void)out_size; (void)ws_size;

  const float* Q  = (const float*)d_in[0];   // [B,S,D]
  const float* K  = (const float*)d_in[1];   // [B,S,D]
  const float* Wq = (const float*)d_in[2];   // [R,D]
  const float* Wk = (const float*)d_in[3];   // [R,D]
  // d_in[4] = k (TOPK), compile-time constant here.

  __bf16* Qa  = (__bf16*)d_ws;                           // bf16 [B,S,R]   2 MiB
  __bf16* Ka  = Qa  + (size_t)B_ * S_ * R_;              // bf16 [B,S,R]   2 MiB
  __bf16* Wqb = Ka  + (size_t)B_ * S_ * R_;              // bf16 [R,D]   0.5 MiB
  __bf16* Wkb = Wqb + (size_t)R_ * D_;                   // bf16 [R,D]   0.5 MiB

  li_cvt_w_kernel<<<dim3(256), dim3(256), 0, stream>>>(Wq, Wk, Wqb, Wkb);

  li_proj_kernel<<<dim3(256), dim3(128), 0, stream>>>(Q, K, Wqb, Wkb, Qa, Ka);

  int*   oI = (int*)d_out;                                     // [B,S,64] int32
  float* oV = (float*)d_out + (size_t)B_ * S_ * TOPK_;         // [B,S,64] f32

  (void)hipFuncSetAttribute((const void*)li_score_topk_kernel,
                            hipFuncAttributeMaxDynamicSharedMemorySize, LDSB);
  li_score_topk_kernel<<<dim3(512), dim3(512), LDSB, stream>>>(Qa, Ka, oI, oV);
}
